// MoE_56719338111431
// MI455X (gfx1250) — compile-verified, matrix operand order checked by reading, not run on hardware
//
#include <hip/hip_runtime.h>
#include <hip/hip_bf16.h>

// Problem constants (from the reference)
#define NTOK 8192   // B*S
#define DDIM 1024
#define ODIM 1024
#define NEXP 8
// top-K = 2

typedef __attribute__((ext_vector_type(16))) __bf16 v16bf;
typedef __attribute__((ext_vector_type(8)))  float  v8f;

// ---------------------------------------------------------------------------
// Kernel 0: zero output accumulator + expert counters
// ---------------------------------------------------------------------------
__global__ __launch_bounds__(256) void moe_zero(float* __restrict__ out,
                                                int* __restrict__ cnt) {
  size_t i = (size_t)blockIdx.x * blockDim.x + threadIdx.x;   // float4 index
  float4 z = {0.f, 0.f, 0.f, 0.f};
  ((float4*)out)[i] = z;
  if (blockIdx.x == 0 && threadIdx.x < NEXP) cnt[threadIdx.x] = 0;
}

// ---------------------------------------------------------------------------
// Kernel 1: We[e][d][o] fp32  ->  WeT[e][o][d] bf16 (column-major weights).
// 64x64 tile per block via padded LDS transpose. One-time 48 MB of traffic.
// ---------------------------------------------------------------------------
__global__ __launch_bounds__(256) void moe_cvt_weT(const float* __restrict__ We,
                                                   __bf16* __restrict__ WeT) {
  const int e  = blockIdx.z;
  const int d0 = blockIdx.x * 64;
  const int o0 = blockIdx.y * 64;
  __shared__ __align__(8) __bf16 tile[64][72];   // pad to dodge bank conflicts

  const int tid = threadIdx.x;
  const float* src = We + (size_t)e * DDIM * ODIM;
#pragma unroll
  for (int p = 0; p < 4; ++p) {                  // 4096 elems / (256 thr * 4)
    const int l = p * 1024 + tid * 4;
    const int d = l >> 6, o = l & 63;
    const float4 v = *(const float4*)(src + (size_t)(d0 + d) * ODIM + o0 + o);
    tile[o + 0][d] = (__bf16)v.x;
    tile[o + 1][d] = (__bf16)v.y;
    tile[o + 2][d] = (__bf16)v.z;
    tile[o + 3][d] = (__bf16)v.w;
  }
  __syncthreads();
  __bf16* dst = WeT + (size_t)e * ODIM * DDIM;
#pragma unroll
  for (int p = 0; p < 4; ++p) {
    const int l = p * 1024 + tid * 4;
    const int oo = l >> 6, dd = l & 63;
    *(uint2*)(dst + (size_t)(o0 + oo) * DDIM + d0 + dd) =
        *(const uint2*)&tile[oo][dd];            // 4 bf16 = 8 B contiguous
  }
}

// ---------------------------------------------------------------------------
// Kernel 2: gating — softmax(x@Wg + bg), top-2, renormalize, scatter to
// per-expert slot lists. One wave32 per token.
// ---------------------------------------------------------------------------
__global__ __launch_bounds__(256) void moe_gate(const float* __restrict__ x,
                                                const float* __restrict__ Wg,
                                                const float* __restrict__ bg,
                                                int*   __restrict__ cnt,
                                                int*   __restrict__ slot_tok,
                                                float* __restrict__ slot_w) {
  const int lane = threadIdx.x & 31;
  const int n = blockIdx.x * 8 + (threadIdx.x >> 5);   // token id (8 waves/block)

  float acc[NEXP];
#pragma unroll
  for (int e = 0; e < NEXP; ++e) acc[e] = 0.f;

  for (int d = lane; d < DDIM; d += 32) {
    const float xv = x[(size_t)n * DDIM + d];
    const float4 w0 = *(const float4*)(Wg + (size_t)d * NEXP);
    const float4 w1 = *(const float4*)(Wg + (size_t)d * NEXP + 4);
    acc[0] += xv * w0.x; acc[1] += xv * w0.y;
    acc[2] += xv * w0.z; acc[3] += xv * w0.w;
    acc[4] += xv * w1.x; acc[5] += xv * w1.y;
    acc[6] += xv * w1.z; acc[7] += xv * w1.w;
  }
#pragma unroll
  for (int off = 16; off > 0; off >>= 1)
#pragma unroll
    for (int e = 0; e < NEXP; ++e) acc[e] += __shfl_xor(acc[e], off, 32);

  if (lane == 0) {
    float lg[NEXP];
#pragma unroll
    for (int e = 0; e < NEXP; ++e) lg[e] = acc[e] + bg[e];
    int i0 = 0;
#pragma unroll
    for (int e = 1; e < NEXP; ++e) if (lg[e] > lg[i0]) i0 = e;
    int i1 = (i0 == 0) ? 1 : 0;
#pragma unroll
    for (int e = 0; e < NEXP; ++e) if (e != i0 && lg[e] > lg[i1]) i1 = e;
    const float g1 = __expf(lg[i1] - lg[i0]);   // g0 = 1, softmax denom cancels
    const float inv = 1.f / (1.f + g1);

    int p0 = atomicAdd(&cnt[i0], 1);
    slot_tok[i0 * NTOK + p0] = n;  slot_w[i0 * NTOK + p0] = inv;
    int p1 = atomicAdd(&cnt[i1], 1);
    slot_tok[i1 * NTOK + p1] = n;  slot_w[i1 * NTOK + p1] = g1 * inv;
  }
}

// ---------------------------------------------------------------------------
// Kernel 3: grouped bf16-WMMA GEMM over routed tokens.
// grid: (token_tile up to 512, O/256 = 4, E = 8), 256 threads = 8 waves.
// A: 16 gathered token rows x 1024 staged to LDS as bf16 ONCE (32 KB, shared
//    by all 8 waves). B: per-lane 32 B contiguous fragments straight from the
//    precomputed column-major bf16 WeT (L2-resident). Each wave owns two
//    16x16 output tiles sharing one A fragment -> 2 WMMAs per K-chunk.
// ---------------------------------------------------------------------------
__global__ __launch_bounds__(256) void moe_gemm(const float* __restrict__ x,
                                                const __bf16* __restrict__ WeT,
                                                const float* __restrict__ be,
                                                const int*   __restrict__ cnt,
                                                const int*   __restrict__ slot_tok,
                                                const float* __restrict__ slot_w,
                                                float* __restrict__ out) {
  const int e    = blockIdx.z;
  const int tile = blockIdx.x;           // 16-token tile within expert list
  const int o0   = blockIdx.y * 256;     // O-tile base (8 waves * 2 accs * 16)
  const int c    = cnt[e];
  if (tile * 16 >= c) return;            // early-exit past routed count

  __shared__ __align__(32) __bf16 ldsA[16 * DDIM];   // 32 KB
  __shared__ int   s_tok[16];
  __shared__ float s_w[16];

  const int tid = threadIdx.x;
  if (tid < 16) {
    int s = tile * 16 + tid;
    int tok = 0; float w = 0.f;
    if (s < c) { tok = slot_tok[e * NTOK + s]; w = slot_w[e * NTOK + s]; }
    s_tok[tid] = tok;  s_w[tid] = w;
  }
  __syncthreads();

  // Stage A once: 16 gathered rows x 1024 fp32 -> bf16
#pragma unroll 4
  for (int r = 0; r < 16; ++r) {
    const float4 v = *(const float4*)(x + (size_t)s_tok[r] * DDIM + tid * 4);
    __bf16* dst = &ldsA[r * DDIM + tid * 4];
    dst[0] = (__bf16)v.x; dst[1] = (__bf16)v.y;
    dst[2] = (__bf16)v.z; dst[3] = (__bf16)v.w;
  }
  __syncthreads();

  const int wave = tid >> 5;             // 0..7
  const int lane = tid & 31;
  const int row  = lane & 15;            // A row (M)
  const int kb   = (lane >> 4) * 16;     // K half: contiguous k = 16*h + s, same for A and B

  // B fragment base pointers: column-major bf16, 32 B contiguous per chunk
  const __bf16* b0p = WeT + ((size_t)e * ODIM + (o0 + wave * 32 + row)) * DDIM + kb;
  const __bf16* b1p = b0p + 16 * DDIM;   // +16 columns

  v8f acc0 = {0.f,0.f,0.f,0.f,0.f,0.f,0.f,0.f};
  v8f acc1 = {0.f,0.f,0.f,0.f,0.f,0.f,0.f,0.f};

#pragma unroll 4
  for (int k0 = 0; k0 < DDIM; k0 += 32) {
    const v16bf a  = *(const v16bf*)&ldsA[row * DDIM + k0 + kb];
    const v16bf b0 = *(const v16bf*)(b0p + k0);
    const v16bf b1 = *(const v16bf*)(b1p + k0);
    acc0 = __builtin_amdgcn_wmma_f32_16x16x32_bf16(false, a, false, b0,
                                                   (short)0, acc0, false, false);
    acc1 = __builtin_amdgcn_wmma_f32_16x16x32_bf16(false, a, false, b1,
                                                   (short)0, acc1, false, false);
  }

  // Epilogue: C/D layout -> lane L, vgpr i holds (m = i or i+8, n = L&15).
  const int ncol = lane & 15;
#pragma unroll
  for (int j = 0; j < 2; ++j) {
    const int oc = o0 + wave * 32 + j * 16 + ncol;
    const float bias = be[e * ODIM + oc];
    const v8f acc = j ? acc1 : acc0;
#pragma unroll
    for (int i = 0; i < 8; ++i) {
      const int m   = (lane < 16) ? i : (8 + i);
      const float w = s_w[m];                 // 0 for padded slots -> adds 0.0
      const int tok = s_tok[m];
      atomicAdd(&out[(size_t)tok * ODIM + oc], w * (acc[i] + bias));
    }
  }
}

// ---------------------------------------------------------------------------
// Launch: zero -> weight transpose/convert -> gate -> grouped WMMA GEMM.
// Workspace layout in d_ws:
//   [0, 16MB)       : __bf16 WeT[E][O][D]   (column-major bf16 weights)
//   [16MB, +256B)   : int cnt[E]
//   [+256B, +256KB) : int   slot_tok[E*N]
//   [..., +256KB)   : float slot_w [E*N]
// ---------------------------------------------------------------------------
extern "C" void kernel_launch(void* const* d_in, const int* in_sizes, int n_in,
                              void* d_out, int out_size, void* d_ws, size_t ws_size,
                              hipStream_t stream) {
  (void)in_sizes; (void)n_in; (void)out_size; (void)ws_size;
  const float* x  = (const float*)d_in[0];
  const float* We = (const float*)d_in[1];
  const float* be = (const float*)d_in[2];
  const float* Wg = (const float*)d_in[3];
  const float* bg = (const float*)d_in[4];
  float* out = (float*)d_out;

  char* ws = (char*)d_ws;
  __bf16* WeT     = (__bf16*)ws;
  const size_t weT_bytes = (size_t)NEXP * ODIM * DDIM * sizeof(__bf16);  // 16 MB
  int*   cnt      = (int*)(ws + weT_bytes);
  int*   slot_tok = (int*)(ws + weT_bytes + 256);
  float* slot_w   = (float*)(ws + weT_bytes + 256 + (size_t)NEXP * NTOK * sizeof(int));

  moe_zero<<<(NTOK * ODIM) / (256 * 4), 256, 0, stream>>>(out, cnt);

  dim3 gt(DDIM / 64, ODIM / 64, NEXP);
  moe_cvt_weT<<<gt, 256, 0, stream>>>(We, WeT);

  moe_gate<<<NTOK / 8, 256, 0, stream>>>(x, Wg, bg, cnt, slot_tok, slot_w);

  dim3 gg(NTOK / 16, ODIM / 256, NEXP);
  moe_gemm<<<gg, 256, 0, stream>>>(x, WeT, be, cnt, slot_tok, slot_w, out);
}
